// Attention_71030169141804
// MI455X (gfx1250) — compile-verified, matrix operand order checked by reading, not run on hardware
//
#include <hip/hip_runtime.h>
#include <hip/hip_bf16.h>

// ---------------------------------------------------------------------------
// GPT-2 attention block for MI455X (gfx1250, wave32, WMMA f32_16x16x32_f16)
//   hidden[2,2048,768] -> qkv -> 12-head causal attention -> proj -> out
// GEMMs: 8 waves/block, each wave computes a 16x64 tile (4 WMMA accumulators,
// A-fragment reused 4x). All 4 B fragments are loaded into distinct registers
// BEFORE the WMMA group so the 4 matrix ops issue back-to-back.
// ---------------------------------------------------------------------------

typedef __attribute__((ext_vector_type(16))) _Float16 v16h;
typedef __attribute__((ext_vector_type(8)))  _Float16 h8;
typedef __attribute__((ext_vector_type(8)))  float    v8f;
typedef __attribute__((ext_vector_type(4)))  float    f4;

#define NX    768
#define NHEAD 12
#define SEQ   2048
#define HD    64
#define BS    2
#define MTOT  (BS * SEQ)          // 4096 rows
#define LDH   40                  // LDS row stride in halves (80B, keeps 16B align)

__device__ __forceinline__ v8f wmma_f16(v16h a, v16h b, v8f c) {
    return __builtin_amdgcn_wmma_f32_16x16x32_f16(false, a, false, b,
                                                  (short)0, c, false, false);
}

__device__ __forceinline__ v16h pack16(h8 lo, h8 hi) {
    v16h r;
#pragma unroll
    for (int j = 0; j < 8; ++j) { r[j] = lo[j]; r[j + 8] = hi[j]; }
    return r;
}

__device__ __forceinline__ h8 cvt8(f4 a, f4 b) {
    h8 r;
#pragma unroll
    for (int j = 0; j < 4; ++j) { r[j] = (_Float16)a[j]; r[j + 4] = (_Float16)b[j]; }
    return r;
}

// ---------------------------------------------------------------------------
// Kernel 1: qkv = X @ W + b ; scatter columns to contiguous f16 Q/K/V buffers
// X:[4096,768] f32, W:[768,2304] f32, bias:[2304] f32
// grid: (2304/64, 4096/128), block 256: wave w owns rows [m0+16w, m0+16w+16)
// ---------------------------------------------------------------------------
__global__ __launch_bounds__(256)
void qkv_gemm_kernel(const float* __restrict__ X, const float* __restrict__ W,
                     const float* __restrict__ bias,
                     _Float16* __restrict__ Q, _Float16* __restrict__ K,
                     _Float16* __restrict__ V)
{
    __shared__ _Float16 ldsB[64 * LDH];           // B tile, transposed [n][k]

    const int t    = threadIdx.x;
    const int wave = t >> 5, lane = t & 31;
    const int lm   = lane & 15, lh = lane >> 4;
    const int m0   = blockIdx.y * 128 + wave * 16;
    const int n0   = blockIdx.x * 64;
    const int arow = m0 + lm;                     // A-fragment row for this lane

    v8f acc[4] = {};

    for (int k0 = 0; k0 < NX; k0 += 32) {
        __syncthreads();
        // stage W[k0..k0+32)[n0..n0+64) transposed into LDS as f16
#pragma unroll
        for (int i = 0; i < 8; ++i) {
            int idx = i * 256 + t;
            int kk = idx >> 6, nn = idx & 63;
            ldsB[nn * LDH + kk] = (_Float16)W[(size_t)(k0 + kk) * 2304 + n0 + nn];
        }
        // prefetch next K-step's W tile (64 cachelines) while we compute
        if (k0 + 32 < NX && t < 64)
            __builtin_prefetch(W + (size_t)(k0 + 32 + (t >> 1)) * 2304 +
                                   n0 + (t & 1) * 32, 0, 3);
        __syncthreads();

        // A fragment: per-lane contiguous runs from X (f32 -> f16), reused 4x
        const float* pa = X + (size_t)arow * NX + k0 + lh * 8;
        h8 alo = cvt8(*(const f4*)(pa),      *(const f4*)(pa + 4));
        h8 ahi = cvt8(*(const f4*)(pa + 16), *(const f4*)(pa + 20));
        v16h a = pack16(alo, ahi);

        // gather all 4 B fragments first, then fire WMMAs back-to-back
        v16h bf[4];
#pragma unroll
        for (int nt = 0; nt < 4; ++nt) {
            const _Float16* pb = ldsB + (nt * 16 + lm) * LDH + lh * 8;
            bf[nt] = pack16(*(const h8*)(pb), *(const h8*)(pb + 16));
        }
#pragma unroll
        for (int nt = 0; nt < 4; ++nt)
            acc[nt] = wmma_f16(a, bf[nt], acc[nt]);
    }

    // epilogue: 64-wide N-tile never straddles the Q/K/V boundary (all mult of 64)
    _Float16* dst; int c0;
    if (n0 < NX)          { dst = Q; c0 = n0; }
    else if (n0 < 2 * NX) { dst = K; c0 = n0 - NX; }
    else                  { dst = V; c0 = n0 - 2 * NX; }
#pragma unroll
    for (int nt = 0; nt < 4; ++nt) {
        const float bv = bias[n0 + nt * 16 + lm];
#pragma unroll
        for (int i = 0; i < 8; ++i) {
            int r = m0 + i + 8 * lh;              // C-layout row
            dst[(size_t)r * NX + c0 + nt * 16 + lm] = (_Float16)(acc[nt][i] + bv);
        }
    }
}

// ---------------------------------------------------------------------------
// Kernel 2: per-head causal attention with online softmax.
// Q/K/V: contiguous f16, head (b,h) at b*SEQ*NX + h*SEQ*HD, row stride 64.
// grid: (SEQ/128, BS*NHEAD), block 256: each wave owns 16 query rows.
// ---------------------------------------------------------------------------
__global__ __launch_bounds__(256)
void attn_kernel(const _Float16* __restrict__ Q, const _Float16* __restrict__ K,
                 const _Float16* __restrict__ V, const float* __restrict__ AM,
                 _Float16* __restrict__ CTX)
{
    __shared__ _Float16 ldsV[64 * LDH];           // V chunk transposed [d][k]
    __shared__ _Float16 ldsP[8 * 16 * LDH];       // per-wave P tile [m][k]

    const int t    = threadIdx.x;
    const int wave = t >> 5, lane = t & 31;
    const int lm   = lane & 15, lh = lane >> 4;
    const int bh   = blockIdx.y;                  // 0..23
    const int b    = bh / NHEAD;
    const size_t base = (size_t)b * SEQ * NX + (size_t)(bh % NHEAD) * SEQ * HD;
    const int q0   = blockIdx.x * 128 + wave * 16;
    const float* amp = AM + (size_t)b * SEQ;

    // Q A-fragments (d = 0..31 and 32..63), loaded once
    const _Float16* qrow = Q + base + (size_t)(q0 + lm) * HD;
    v16h aq[2];
#pragma unroll
    for (int ks = 0; ks < 2; ++ks) {
        const _Float16* p = qrow + ks * 32 + lh * 8;
        aq[ks] = pack16(*(const h8*)(p), *(const h8*)(p + 16));
    }

    v8f o[4] = {};                                // 16q x 64d accumulators
    float mrun[8], lrun[8];
#pragma unroll
    for (int i = 0; i < 8; ++i) { mrun[i] = -1e30f; lrun[i] = 0.0f; }

    const int kc_end = blockIdx.x * 128 + 128;    // causal block bound

    for (int kc0 = 0; kc0 < kc_end; kc0 += 32) {
        __syncthreads();
        // stage V[kc0..kc0+32)[0..64) transposed into LDS
        {
            int r = t >> 3, c = (t & 7) * 8;
            h8 v = *(const h8*)(V + base + (size_t)(kc0 + r) * HD + c);
#pragma unroll
            for (int j = 0; j < 8; ++j) ldsV[(c + j) * LDH + r] = v[j];
        }
        // prefetch next V chunk (32 rows = 32 cachelines)
        if (kc0 + 32 < kc_end && t < 32)
            __builtin_prefetch(V + base + (size_t)(kc0 + 32 + t) * HD, 0, 3);
        __syncthreads();

        const bool active = (kc0 <= q0 + 15);     // wave-uniform
        if (active) {
            // scores: two 16x16 tiles over this 32-key chunk.
            // Load all 4 K fragments first; interleave WMMAs across the two
            // independent accumulators so they issue back-to-back.
            v16h bk[2][2];
#pragma unroll
            for (int tile = 0; tile < 2; ++tile) {
                const _Float16* krow = K + base + (size_t)(kc0 + tile * 16 + lm) * HD;
#pragma unroll
                for (int ks = 0; ks < 2; ++ks) {
                    const _Float16* p = krow + ks * 32 + lh * 8;
                    bk[tile][ks] = pack16(*(const h8*)(p), *(const h8*)(p + 16));
                }
            }
            v8f s[2] = { {}, {} };
#pragma unroll
            for (int ks = 0; ks < 2; ++ks) {
#pragma unroll
                for (int tile = 0; tile < 2; ++tile)
                    s[tile] = wmma_f16(aq[ks], bk[tile][ks], s[tile]);
            }
            // mask + online softmax (rows live in 16-lane halves)
            const int col0 = kc0 + lm, col1 = col0 + 16;
            const float am0 = amp[col0], am1 = amp[col1];
            _Float16* pp = ldsP + wave * 16 * LDH;
#pragma unroll
            for (int i = 0; i < 8; ++i) {
                const int qr = q0 + i + 8 * lh;
                float s0 = (col0 <= qr) ? s[0][i] * 0.125f : -10000.0f;
                float s1 = (col1 <= qr) ? s[1][i] * 0.125f : -10000.0f;
                s0 += am0; s1 += am1;
                float mx = fmaxf(s0, s1);
#pragma unroll
                for (int off = 1; off < 16; off <<= 1)
                    mx = fmaxf(mx, __shfl_xor(mx, off, 32));
                const float mn = fmaxf(mrun[i], mx);
                const float sc = __expf(mrun[i] - mn);
                const float e0 = __expf(s0 - mn), e1 = __expf(s1 - mn);
                float rs = e0 + e1;
#pragma unroll
                for (int off = 1; off < 16; off <<= 1)
                    rs += __shfl_xor(rs, off, 32);
                lrun[i] = lrun[i] * sc + rs;
                mrun[i] = mn;
                const int r = i + 8 * lh;
                pp[r * LDH + lm]      = (_Float16)e0;
                pp[r * LDH + 16 + lm] = (_Float16)e1;
                o[0][i] *= sc; o[1][i] *= sc; o[2][i] *= sc; o[3][i] *= sc;
            }
        }
        __syncthreads();
        if (active) {
            // P (A-layout) @ V (B-layout from transposed LDS):
            // load all 4 V fragments, then 4 independent-accumulator WMMAs.
            const _Float16* ppr = ldsP + wave * 16 * LDH + lm * LDH + lh * 8;
            v16h ap = pack16(*(const h8*)(ppr), *(const h8*)(ppr + 16));
            v16h bv[4];
#pragma unroll
            for (int nt = 0; nt < 4; ++nt) {
                const _Float16* pv = ldsV + (nt * 16 + lm) * LDH + lh * 8;
                bv[nt] = pack16(*(const h8*)(pv), *(const h8*)(pv + 16));
            }
#pragma unroll
            for (int nt = 0; nt < 4; ++nt)
                o[nt] = wmma_f16(ap, bv[nt], o[nt]);
        }
    }

    // normalize and store context as f16: [bh][q][d] contiguous
#pragma unroll
    for (int i = 0; i < 8; ++i) {
        const float inv = 1.0f / lrun[i];
        const int qr = q0 + i + 8 * lh;
#pragma unroll
        for (int nt = 0; nt < 4; ++nt)
            CTX[((size_t)bh * SEQ + qr) * HD + nt * 16 + lm] =
                (_Float16)(o[nt][i] * inv);
    }
}

// ---------------------------------------------------------------------------
// Kernel 3: out = ctx_perm @ Wp + bp  (transpose(0,2,1,3) folded into A loads)
// CTX: f16 [bs,12,2048,64]; Wp:[768,768] f32; out f32 [4096,768]
// grid: (768/64, 4096/128), block 256, wave tile 16x64
// ---------------------------------------------------------------------------
__global__ __launch_bounds__(256)
void proj_gemm_kernel(const _Float16* __restrict__ CTX, const float* __restrict__ W,
                      const float* __restrict__ bias, float* __restrict__ OUT)
{
    __shared__ _Float16 ldsB[64 * LDH];

    const int t    = threadIdx.x;
    const int wave = t >> 5, lane = t & 31;
    const int lm   = lane & 15, lh = lane >> 4;
    const int m0   = blockIdx.y * 128 + wave * 16;
    const int n0   = blockIdx.x * 64;
    const int arow = m0 + lm;
    const int bb   = arow >> 11, ss = arow & (SEQ - 1);

    v8f acc[4] = {};

    for (int k0 = 0; k0 < NX; k0 += 32) {
        __syncthreads();
#pragma unroll
        for (int i = 0; i < 8; ++i) {
            int idx = i * 256 + t;
            int kk = idx >> 6, nn = idx & 63;
            ldsB[nn * LDH + kk] = (_Float16)W[(size_t)(k0 + kk) * NX + n0 + nn];
        }
        if (k0 + 32 < NX && t < 64)
            __builtin_prefetch(W + (size_t)(k0 + 32 + (t >> 1)) * NX +
                                   n0 + (t & 1) * 32, 0, 3);
        __syncthreads();

        // A fragment: channel c = h*64+d maps to CTX[b][h][s][d]; reused 4x
        const int h  = k0 >> 6;                   // 32-chunk stays inside one head
        const int d0 = k0 & 63;
        const _Float16* ar =
            CTX + ((size_t)(bb * NHEAD + h) * SEQ + ss) * HD + d0 + lh * 8;
        v16h a = pack16(*(const h8*)(ar), *(const h8*)(ar + 16));

        v16h bf[4];
#pragma unroll
        for (int nt = 0; nt < 4; ++nt) {
            const _Float16* pb = ldsB + (nt * 16 + lm) * LDH + lh * 8;
            bf[nt] = pack16(*(const h8*)(pb), *(const h8*)(pb + 16));
        }
#pragma unroll
        for (int nt = 0; nt < 4; ++nt)
            acc[nt] = wmma_f16(a, bf[nt], acc[nt]);
    }

#pragma unroll
    for (int nt = 0; nt < 4; ++nt) {
        const int ncol = n0 + nt * 16 + lm;
        const float bv = bias[ncol];
#pragma unroll
        for (int i = 0; i < 8; ++i) {
            int r = m0 + i + 8 * lh;
            OUT[(size_t)r * NX + ncol] = acc[nt][i] + bv;
        }
    }
}

// ---------------------------------------------------------------------------
extern "C" void kernel_launch(void* const* d_in, const int* in_sizes, int n_in,
                              void* d_out, int out_size, void* d_ws, size_t ws_size,
                              hipStream_t stream)
{
    const float* hs     = (const float*)d_in[0];   // [2,2048,768]
    const float* amask  = (const float*)d_in[1];   // [2,1,1,2048]
    const float* w_attn = (const float*)d_in[2];   // [768,2304]
    const float* b_attn = (const float*)d_in[3];   // [2304]
    const float* w_proj = (const float*)d_in[4];   // [768,768]
    const float* b_proj = (const float*)d_in[5];   // [768]
    float* out = (float*)d_out;

    const size_t nelem = (size_t)MTOT * NX;        // 3,145,728 halves per buffer
    _Float16* Q   = (_Float16*)d_ws;
    _Float16* K   = Q + nelem;
    _Float16* V   = K + nelem;
    _Float16* CTX = V + nelem;

    qkv_gemm_kernel<<<dim3(2304 / 64, MTOT / 128), 256, 0, stream>>>(
        hs, w_attn, b_attn, Q, K, V);
    attn_kernel<<<dim3(SEQ / 128, BS * NHEAD), 256, 0, stream>>>(
        Q, K, V, amask, CTX);
    proj_gemm_kernel<<<dim3(NX / 64, MTOT / 128), 256, 0, stream>>>(
        CTX, w_proj, b_proj, out);
}
